// MultiHeadCrossSimilarity_67362267070939
// MI455X (gfx1250) — matrix operand order repaired and checked by reading, not compiled, not run on hardware
//
#include <hip/hip_runtime.h>
#include <hip/hip_bf16.h>

// ---------------------------------------------------------------------------
// MultiHeadCrossSimilarity on MI455X (gfx1250), wave32 + WMMA bf16.
//
//   1) cvt:      f32 -> bf16 for x, Wq, Wk
//   2) proj:     Q = x*Wq^T + bq ; K = x*Wk^T + bk   (bf16 out, head-split)
//   3) simstats: block = (8 query tiles, partner group j, head h).
//                K-group B chunks (128 boxes x 32 d) staged into LDS with
//                CDNA5 async global->LDS DMA (GLOBAL_LOAD_ASYNC_TO_LDS_B128,
//                ASYNCcnt), double-buffered, shared by all 8 waves.
//                Per wave: 8 accumulators, B fragments via ds_load_b128.
//   4) finalize: cross-group (j != g) avg/max/min -> [8192, 18] f32
// ---------------------------------------------------------------------------

typedef __attribute__((ext_vector_type(16))) __bf16        v16bf;
typedef __attribute__((ext_vector_type(8)))  float         v8f;
typedef __attribute__((ext_vector_type(4)))  unsigned int  u32x4;

union FragBF {
    v16bf bf;
    u32x4 q[2];
};

__device__ __forceinline__ unsigned short f2bf(float f) {
    unsigned int u = __float_as_uint(f);
    unsigned int r = u + 0x7fffu + ((u >> 16) & 1u);   // round-to-nearest-even
    return (unsigned short)(r >> 16);
}

// A 16x32 bf16 fragment: lanes 0-15 hold K 0..15 of row (lane&15),
// lanes 16-31 hold K 16..31.
__device__ __forceinline__ void load_a(FragBF& f, const unsigned short* pa,
                                       int k, int sel) {
    f.q[0] = *(const u32x4*)(pa + k + sel * 16);
    f.q[1] = *(const u32x4*)(pa + k + sel * 16 + 8);
}
// B 32x16 bf16 fragment from global: lanes 0-15 hold K 0..15 of column
// (lane&15), lanes 16-31 hold K 16..31.
__device__ __forceinline__ void load_b(FragBF& f, const unsigned short* pb,
                                       int k, int sel) {
    f.q[0] = *(const u32x4*)(pb + k + sel * 16);
    f.q[1] = *(const u32x4*)(pb + k + sel * 16 + 8);
}

// CDNA5 async DMA: copy 32 contiguous bytes global -> LDS (per lane),
// tracked by ASYNCcnt. Instruction offset applies to both LDS and global
// addresses (ISA 15.18.3), so one base pair + offset:16 covers both pieces.
__device__ __forceinline__ void async_copy_32B(unsigned lds_off,
                                               const unsigned short* gptr) {
    asm volatile(
        "global_load_async_to_lds_b128 %0, %1, off\n\t"
        "global_load_async_to_lds_b128 %0, %1, off offset:16"
        :: "v"(lds_off), "v"(gptr) : "memory");
}
__device__ __forceinline__ void wait_async0() {
    asm volatile("s_wait_asynccnt 0x0" ::: "memory");
}

// ---------------------------------------------------------------- stage 1 ---
__global__ void cvt_f32_bf16(const float* __restrict__ src,
                             unsigned short* __restrict__ dst, int n) {
    int i = blockIdx.x * blockDim.x + threadIdx.x;
    int stride = gridDim.x * blockDim.x;
    for (; i < n; i += stride) dst[i] = f2bf(src[i]);
}

// ---------------------------------------------------------------- stage 2 ---
// One wave per 16x64 output slab (4 accumulators). W is L2-resident.
__global__ __launch_bounds__(256) void proj_kernel(
    const unsigned short* __restrict__ xb,
    const unsigned short* __restrict__ Wqb,
    const unsigned short* __restrict__ Wkb,
    const float* __restrict__ bq,
    const float* __restrict__ bk,
    unsigned short* __restrict__ Qh,
    unsigned short* __restrict__ Kh) {
    const int lane = threadIdx.x & 31;
    const int wave = threadIdx.x >> 5;
    const int w    = blockIdx.x * 8 + wave;      // 0..16383
    const bool isK = (w >= 8192);
    const int t    = w & 8191;
    const int tm   = t >> 4;                     // 0..511 (16-row tile)
    const int tsn  = t & 15;                     // 0..15  (64-col supertile)

    const unsigned short* W    = isK ? Wkb : Wqb;
    const float*          bias = isK ? bk  : bq;
    unsigned short*       Out  = isK ? Kh  : Qh;

    const int arow = lane & 15;
    const int sel  = lane >> 4;                  // lane half (0/1)
    const unsigned short* pa = xb + (size_t)(tm * 16 + arow) * 1024;
    const int e0 = tsn * 64 + (lane & 15);       // B column for c=0
    const unsigned short* pb = W + (size_t)e0 * 1024;  // c offsets fold to imm

    v8f acc[4] = {};
    for (int k = 0; k < 1024; k += 32) {
        FragBF a;
        load_a(a, pa, k, sel);
        #pragma unroll
        for (int c = 0; c < 4; ++c) {
            FragBF b;
            load_b(b, pb + c * 16 * 1024, k, sel);
            acc[c] = __builtin_amdgcn_wmma_f32_16x16x32_bf16(
                false, a.bf, false, b.bf, (short)0, acc[c], false, false);
        }
    }

    const int mb = sel * 8;                      // C layout: VGPR r -> row mb+r
    #pragma unroll
    for (int c = 0; c < 4; ++c) {
        const int e     = e0 + c * 16;
        const float bv  = bias[e];
        const int  h    = e >> 9;                // head split of 1024 features
        const int  dcol = e & 511;
        #pragma unroll
        for (int r = 0; r < 8; ++r) {
            int row = tm * 16 + mb + r;
            Out[((size_t)h * 8192 + row) * 512 + dcol] = f2bf(acc[c][r] + bv);
        }
    }
}

// ---------------------------------------------------------------- stage 3 ---
// grid = (64 qblocks, 16 j, 2 h), block = 256 (8 waves).
// Wave w computes query tile (qblock*8 + w) vs all 512 boxes of group j.
// B chunks (128 boxes x 32 d) staged in LDS via async DMA, double-buffered,
// padded rows (40 ushorts = 80B stride -> conflict-free ds_load_b128).
#define SIM_ROWPAD 40
__global__ __launch_bounds__(256) void sim_stats_kernel(
    const unsigned short* __restrict__ Qh,
    const unsigned short* __restrict__ Kh,
    float* __restrict__ stats) {
    __shared__ unsigned short sB[2][128 * SIM_ROWPAD];   // 2 x 10 KB

    const int tid   = threadIdx.x;
    const int lane  = tid & 31;
    const int wave  = tid >> 5;                  // 0..7
    const int qtile = blockIdx.x * 8 + wave;     // 0..511
    const int j     = blockIdx.y;                // partner group 0..15
    const int h     = blockIdx.z;                // head 0..1
    const int sel   = lane >> 4;
    const int arow  = lane & 15;

    const unsigned short* pa = Qh + ((size_t)h * 8192 + qtile * 16 + arow) * 512;
    const unsigned short* Kg = Kh + ((size_t)h * 8192 + j * 512) * 512;

    // staging: thread tid copies 32B (row srow, 16B pieces sj0, sj0+1)
    const int srow = tid >> 1;                   // 0..127
    const int sj0  = (tid & 1) * 2;              // 0 or 2 (of 4 pieces/row)
    const unsigned so0 =
        (unsigned)(size_t)&sB[0][srow * SIM_ROWPAD + sj0 * 8];
    const unsigned so1 =
        (unsigned)(size_t)&sB[1][srow * SIM_ROWPAD + sj0 * 8];

    float rsum[8], rmax[8], rmin[8];
    #pragma unroll
    for (int r = 0; r < 8; ++r) {
        rsum[r] = 0.0f; rmax[r] = -3.4e38f; rmin[r] = 3.4e38f;
    }

    // prologue: async-stage chunk s=0 (p=0, kk=0) into buffer 0
    async_copy_32B(so0, Kg + (size_t)srow * 512 + sj0 * 8);

    v8f acc[8] = {};
    // 64 chunk-steps: s = p*16 + kk  (p: 128-box slab, kk: 32-d step)
    for (int s = 0; s < 64; ++s) {
        const int kk  = s & 15;
        const int cur = s & 1;

        wait_async0();         // own async writes for chunk s have landed
        __syncthreads();       // all waves' writes landed; all prior reads of
                               // the other buffer (step s-1) are complete

        // issue async staging for chunk s+1 into the other buffer;
        // it overlaps the entire compute phase below
        if (s < 63) {
            const int pn = (s + 1) >> 4;
            const int kn = ((s + 1) & 15) * 32;
            const unsigned short* g =
                Kg + (size_t)(pn * 128 + srow) * 512 + kn + sj0 * 8;
            async_copy_32B(cur ? so0 : so1, g);
        }

        // A fragment (direct global, L0/L2-resident; 2 loads)
        FragBF a;
        load_a(a, pa, kk * 32, sel);

        // 8 WMMAs fed from LDS
        const unsigned short* sbuf = sB[cur];
        #pragma unroll
        for (int c = 0; c < 8; ++c) {
            FragBF b;
            const unsigned short* pb =
                sbuf + (c * 16 + arow) * SIM_ROWPAD + sel * 16;
            b.q[0] = *(const u32x4*)(pb);
            b.q[1] = *(const u32x4*)(pb + 8);
            acc[c] = __builtin_amdgcn_wmma_f32_16x16x32_bf16(
                false, a.bf, false, b.bf, (short)0, acc[c], false, false);
        }

        // end of a 512-d accumulation (one 128-box slab done): fold stats
        if (kk == 15) {
            #pragma unroll
            for (int c = 0; c < 8; ++c) {
                #pragma unroll
                for (int r = 0; r < 8; ++r) {
                    rsum[r] += acc[c][r];
                    rmax[r] = fmaxf(rmax[r], acc[c][r]);
                    rmin[r] = fminf(rmin[r], acc[c][r]);
                }
            }
            #pragma unroll
            for (int c = 0; c < 8; ++c) acc[c] = (v8f){};
        }
    }

    // reduce across the 16 lanes holding one row-half (wave32 butterfly)
    #pragma unroll
    for (int m = 1; m < 16; m <<= 1) {
        #pragma unroll
        for (int r = 0; r < 8; ++r) {
            rsum[r] += __shfl_xor(rsum[r], m, 32);
            rmax[r] = fmaxf(rmax[r], __shfl_xor(rmax[r], m, 32));
            rmin[r] = fminf(rmin[r], __shfl_xor(rmin[r], m, 32));
        }
    }

    if ((lane & 15) == 0) {
        const int mb = sel * 8;
        #pragma unroll
        for (int r = 0; r < 8; ++r) {
            int qrow = qtile * 16 + mb + r;
            size_t base = (((size_t)qrow * 16 + j) * 2 + h) * 3;
            stats[base + 0] = rsum[r] * (1.0f / 512.0f);  // mean over boxes
            stats[base + 1] = rmax[r];
            stats[base + 2] = rmin[r];
        }
    }
}

// ---------------------------------------------------------------- stage 4 ---
__global__ void finalize_kernel(const float* __restrict__ stats,
                                float* __restrict__ out) {
    int t = blockIdx.x * blockDim.x + threadIdx.x;
    if (t >= 8192 * 2) return;
    const int qrow = t >> 1;
    const int h    = t & 1;
    const int g    = qrow >> 9;                  // own group (512 boxes/group)

    float s[3]  = {0.f, 0.f, 0.f};
    float mx[3] = {-3.4e38f, -3.4e38f, -3.4e38f};
    float mn[3] = {3.4e38f, 3.4e38f, 3.4e38f};
    for (int j = 0; j < 16; ++j) {
        if (j == g) continue;
        size_t base = (((size_t)qrow * 16 + j) * 2 + h) * 3;
        #pragma unroll
        for (int m = 0; m < 3; ++m) {
            float v = stats[base + m];
            s[m] += v;
            mx[m] = fmaxf(mx[m], v);
            mn[m] = fminf(mn[m], v);
        }
    }
    #pragma unroll
    for (int m = 0; m < 3; ++m) {
        out[(size_t)qrow * 18 + (0 * 3 + m) * 2 + h] = s[m] * (1.0f / 15.0f);
        out[(size_t)qrow * 18 + (1 * 3 + m) * 2 + h] = mx[m];
        out[(size_t)qrow * 18 + (2 * 3 + m) * 2 + h] = mn[m];
    }
}

// ---------------------------------------------------------------------------
extern "C" void kernel_launch(void* const* d_in, const int* in_sizes, int n_in,
                              void* d_out, int out_size, void* d_ws, size_t ws_size,
                              hipStream_t stream) {
    const float* x  = (const float*)d_in[0];   // [1, 8192, 1024]
    const float* Wq = (const float*)d_in[1];   // [1024, 1024]
    const float* bq = (const float*)d_in[2];   // [1024]
    const float* Wk = (const float*)d_in[3];   // [1024, 1024]
    const float* bk = (const float*)d_in[4];   // [1024]
    float* out = (float*)d_out;                // [8192, 18]

    size_t off = 0;
    char* base = (char*)d_ws;
    auto carve = [&](size_t bytes) -> void* {
        void* p = base + off;
        off += (bytes + 255) & ~(size_t)255;
        return p;
    };
    unsigned short* xb    = (unsigned short*)carve((size_t)8192 * 1024 * 2);    // 16 MB
    unsigned short* Wqb   = (unsigned short*)carve((size_t)1024 * 1024 * 2);    //  2 MB
    unsigned short* Wkb   = (unsigned short*)carve((size_t)1024 * 1024 * 2);    //  2 MB
    unsigned short* Qh    = (unsigned short*)carve((size_t)2 * 8192 * 512 * 2); // 16 MB
    unsigned short* Kh    = (unsigned short*)carve((size_t)2 * 8192 * 512 * 2); // 16 MB
    float*          stats = (float*)carve((size_t)8192 * 16 * 2 * 3 * 4);       //  3 MB

    cvt_f32_bf16<<<2048, 256, 0, stream>>>(x,  xb,  8192 * 1024);
    cvt_f32_bf16<<<512,  256, 0, stream>>>(Wq, Wqb, 1024 * 1024);
    cvt_f32_bf16<<<512,  256, 0, stream>>>(Wk, Wkb, 1024 * 1024);

    proj_kernel<<<2048, 256, 0, stream>>>(xb, Wqb, Wkb, bq, bk, Qh, Kh);

    sim_stats_kernel<<<dim3(64, 16, 2), 256, 0, stream>>>(Qh, Kh, stats);

    finalize_kernel<<<64, 256, 0, stream>>>(stats, out);
}
